// TritonQwen3Attention_83691732729867
// MI455X (gfx1250) — matrix operand order repaired and checked by reading, not compiled, not run on hardware
//
#include <hip/hip_runtime.h>
#include <hip/hip_bf16.h>

// ---------------------------------------------------------------------------
// Qwen3-style attention block for MI455X (gfx1250), bf16 WMMA throughout.
// ---------------------------------------------------------------------------

typedef __attribute__((ext_vector_type(16))) __bf16 v16bf;
typedef __attribute__((ext_vector_type(8)))  float  v8f;
typedef __attribute__((ext_vector_type(4)))  int    v4i;

union Frag {
    v16bf v;
    uint4 u[2];
};

static constexpr int Bc   = 2;
static constexpr int Sc   = 2048;
static constexpr int HIDc = 1024;
static constexpr int Hc   = 16;
static constexpr int KVc  = 8;
static constexpr int Dc   = 128;
static constexpr float SCALEc = 0.08838834764831845f;  // 128^-0.5
static constexpr float LOG_THETA_OVER_HALF = 0.2158673598f; // ln(1e6)/64

// ---------------------------------------------------------------------------
// CDNA5 async global->LDS DMA path (GLOBAL_LOAD_ASYNC_TO_LDS_B128).
// Builtin signature (per hipcc diagnostic): (v4i AS1*, v4i AS3*, imm, imm).
// Guarded: falls back to global_load + ds_store staging if builtin is absent.
// ---------------------------------------------------------------------------
#if defined(__has_builtin)
#if __has_builtin(__builtin_amdgcn_global_load_async_to_lds_b128)
#define USE_ASYNC_LDS 1
#endif
#endif
#ifndef USE_ASYNC_LDS
#define USE_ASYNC_LDS 0
#endif

#if USE_ASYNC_LDS
typedef __attribute__((address_space(1))) v4i gv4i;
typedef __attribute__((address_space(3))) v4i lv4i;
#endif

static __device__ inline void stage16(__bf16* lptr, const __bf16* gptr) {
#if USE_ASYNC_LDS
    __builtin_amdgcn_global_load_async_to_lds_b128((gv4i*)gptr, (lv4i*)lptr, 0, 0);
#else
    *(uint4*)lptr = *(const uint4*)gptr;
#endif
}

static __device__ inline void wait_async_lds() {
#if USE_ASYNC_LDS
#if __has_builtin(__builtin_amdgcn_s_wait_asynccnt)
    __builtin_amdgcn_s_wait_asynccnt(0);
#else
    asm volatile("s_wait_asynccnt 0x0" ::: "memory");
#endif
#endif
}

static __device__ inline unsigned short f2bf_bits(float f) {
    __bf16 b = (__bf16)f;
    return __builtin_bit_cast(unsigned short, b);
}

static __device__ inline v8f wmma_bf16(const Frag& a, const Frag& b, v8f c) {
    return __builtin_amdgcn_wmma_f32_16x16x32_bf16(false, a.v, false, b.v,
                                                   (short)0, c, false, false);
}

// ---------------------------------------------------------------------------
// fp32 -> bf16 conversion (float4 in, 4x bf16 out)
// ---------------------------------------------------------------------------
__global__ void cvt_f32_bf16(const float4* __restrict__ src,
                             ushort4* __restrict__ dst, int n4) {
    int i = blockIdx.x * blockDim.x + threadIdx.x;
    int stride = gridDim.x * blockDim.x;
    for (; i < n4; i += stride) {
        float4 f = src[i];
        ushort4 o;
        o.x = f2bf_bits(f.x);
        o.y = f2bf_bits(f.y);
        o.z = f2bf_bits(f.z);
        o.w = f2bf_bits(f.w);
        dst[i] = o;
    }
}

// ---------------------------------------------------------------------------
// C[M,N] = A[M,K] * W[N,K]^T   (bf16 inputs, fp32 out)
// block tile 128(M) x 128(N), BK=32, 256 threads = 8 waves (4x2).
// Each wave -> 32(M) x 64(N): 2 A-frags x 4 B-frags = 8 WMMA per K-chunk,
// fragments batch-loaded before the WMMA burst.
// ---------------------------------------------------------------------------
#define GBM 128
#define GBN 128
#define GBK 32

__global__ __launch_bounds__(256) void gemm_bf16_nt(
    const __bf16* __restrict__ A, const __bf16* __restrict__ W,
    float* __restrict__ C, int M, int N, int K) {
    __shared__ __align__(16) __bf16 As[GBM * GBK];
    __shared__ __align__(16) __bf16 Bs[GBN * GBK];

    const int t    = threadIdx.x;
    const int lane = t & 31;
    const int lr   = lane & 15;
    const int half = lane >> 4;
    const int w    = t >> 5;
    const int wm   = w & 3;   // 0..3 -> 32-row M slab
    const int wn   = w >> 2;  // 0..1 -> 64-col N slab
    const int m0   = blockIdx.x * GBM;
    const int n0   = blockIdx.y * GBN;

    v8f acc[2][4] = {};

    for (int kk = 0; kk < K; kk += GBK) {
        __syncthreads();
        // stage A and B tiles (128x32 bf16 each); 4 x 16B chunks per thread
#pragma unroll
        for (int i = 0; i < 2; ++i) {
            int idx = t + 256 * i;
            int r = idx >> 2, sg = idx & 3;
            stage16(&As[r * GBK + sg * 8], &A[(size_t)(m0 + r) * K + kk + sg * 8]);
            stage16(&Bs[r * GBK + sg * 8], &W[(size_t)(n0 + r) * K + kk + sg * 8]);
        }
        wait_async_lds();
        if (kk + GBK < K) {
            __builtin_prefetch(&A[(size_t)(m0 + (t >> 1)) * K + kk + GBK], 0, 3);
            __builtin_prefetch(&W[(size_t)(n0 + (t >> 1)) * K + kk + GBK], 0, 3);
        }
        __syncthreads();

        // batch-load all fragments, then burst 8 WMMAs
        Frag a[2], b[4];
#pragma unroll
        for (int i = 0; i < 2; ++i) {
            const __bf16* ap = &As[(32 * wm + 16 * i + lr) * GBK + 8 * half];
            a[i].u[0] = *(const uint4*)ap;
            a[i].u[1] = *(const uint4*)(ap + 16);
        }
#pragma unroll
        for (int j = 0; j < 4; ++j) {
            const __bf16* bp = &Bs[(64 * wn + 16 * j + lr) * GBK + 8 * half];
            b[j].u[0] = *(const uint4*)bp;
            b[j].u[1] = *(const uint4*)(bp + 16);
        }
#pragma unroll
        for (int i = 0; i < 2; ++i)
#pragma unroll
            for (int j = 0; j < 4; ++j)
                acc[i][j] = wmma_bf16(a[i], b[j], acc[i][j]);
    }

#pragma unroll
    for (int i = 0; i < 2; ++i)
#pragma unroll
        for (int j = 0; j < 4; ++j)
#pragma unroll
            for (int r = 0; r < 8; ++r) {
                int row = m0 + 32 * wm + 16 * i + r + 8 * half;
                int col = n0 + 64 * wn + 16 * j + lr;
                C[(size_t)row * N + col] = acc[i][j][r];
            }
}

// ---------------------------------------------------------------------------
// Per-head RMSNorm + RoPE. One wave per (token, head). D=128 -> 4 elems/lane.
// Optionally writes bf16 [B,nh,S,D], fp32 [B,nh,S,D], bf16-transposed [B,nh,D,S].
// ---------------------------------------------------------------------------
__global__ __launch_bounds__(256) void normrope_kernel(
    const float* __restrict__ in,        // [B*S, nh*128] token-major
    const int* __restrict__ positions,   // [S]
    const float* __restrict__ nw,        // norm weight [128] (may be null)
    __bf16* __restrict__ out_bf,         // [B,nh,S,128] or null
    float* __restrict__ out_f32,         // [B,nh,S,128] or null
    __bf16* __restrict__ out_T,          // [B,nh,128,S] or null
    int nh, int do_rope) {
    int gw   = (blockIdx.x * blockDim.x + threadIdx.x) >> 5;
    int lane = threadIdx.x & 31;
    int tkn  = gw / nh;
    int hh   = gw % nh;
    int b    = tkn / Sc;
    int s    = tkn % Sc;

    const float* ip = in + (size_t)tkn * (nh * Dc) + hh * Dc;
    float vals[4];
#pragma unroll
    for (int j = 0; j < 4; ++j) vals[j] = ip[lane + 32 * j];

    float r[4];
    if (do_rope) {
        float ss = 0.f;
#pragma unroll
        for (int j = 0; j < 4; ++j) ss += vals[j] * vals[j];
#pragma unroll
        for (int off = 16; off > 0; off >>= 1) ss += __shfl_xor(ss, off, 32);
        float rms = rsqrtf(ss * (1.0f / 128.0f) + 1e-6f);

        float n[4], c[4], sn[4];
#pragma unroll
        for (int j = 0; j < 4; ++j) {
            int d = lane + 32 * j;
            n[j] = nw[d] * vals[j] * rms;
            float ang = (float)positions[s] *
                        __expf(-(float)(d & 63) * LOG_THETA_OVER_HALF);
            c[j]  = __cosf(ang);
            sn[j] = __sinf(ang);
        }
        // rotate_half: d<64 pairs with d+64 (j and j+2 within a lane)
        r[0] = n[0] * c[0] - n[2] * sn[0];
        r[1] = n[1] * c[1] - n[3] * sn[1];
        r[2] = n[2] * c[2] + n[0] * sn[2];
        r[3] = n[3] * c[3] + n[1] * sn[3];
    } else {
#pragma unroll
        for (int j = 0; j < 4; ++j) r[j] = vals[j];
    }

    size_t obase = (((size_t)b * nh + hh) * Sc + s) * Dc;
#pragma unroll
    for (int j = 0; j < 4; ++j) {
        int d = lane + 32 * j;
        if (out_bf)  out_bf[obase + d] = (__bf16)r[j];
        if (out_f32) out_f32[obase + d] = r[j];
        if (out_T)
            out_T[(((size_t)b * nh + hh) * Dc + d) * Sc + s] = (__bf16)r[j];
    }
}

// ---------------------------------------------------------------------------
// Flash attention: one wave per 16-query tile per (b,h). 32 keys per step:
// 8 score WMMAs + online softmax + P relayout through 1KB LDS + 8 PV WMMAs.
// Fragments are batch-loaded ahead of each WMMA burst.
// ---------------------------------------------------------------------------
__global__ __launch_bounds__(32) void attn_kernel(
    const __bf16* __restrict__ Q,   // [B,H,S,D]
    const __bf16* __restrict__ Kb,  // [B,KV,S,D]
    const __bf16* __restrict__ VT,  // [B,KV,D,S]
    __bf16* __restrict__ ctx) {     // [B,S,H*D]
    __shared__ __align__(16) __bf16 pl[16 * 32];

    const int qt   = blockIdx.x;
    const int h    = blockIdx.y;
    const int b    = blockIdx.z;
    const int lane = threadIdx.x & 31;
    const int lr   = lane & 15;
    const int half = lane >> 4;
    const int kvh  = h >> 1;  // H/KV == 2

    const __bf16* qb = Q  + (((size_t)b * Hc  + h)   * Sc) * Dc;
    const __bf16* kb = Kb + (((size_t)b * KVc + kvh) * Sc) * Dc;
    const __bf16* vt = VT + (((size_t)b * KVc + kvh) * Dc) * (size_t)Sc;

    // preload 4 Q A-fragments covering D=128
    Frag qf[4];
    const int qrow = qt * 16 + lr;
#pragma unroll
    for (int c = 0; c < 4; ++c) {
        const __bf16* qp = qb + (size_t)qrow * Dc + 32 * c;
        qf[c].u[0] = *(const uint4*)(qp + 8 * half);
        qf[c].u[1] = *(const uint4*)(qp + 16 + 8 * half);
    }

    v8f o[8] = {};
    float m8[8], l8[8];
#pragma unroll
    for (int r = 0; r < 8; ++r) { m8[r] = -1e30f; l8[r] = 0.f; }

    const int nsteps = (qt * 16 + 47) >> 5;  // cover keys 0..qt*16+15
    for (int kts = 0; kts < nsteps; ++kts) {
        const int kbase = kts * 32;

        // ---- scores: batch-load 8 K fragments, then 8 WMMAs ----
        Frag b0[4], b1[4];
#pragma unroll
        for (int c = 0; c < 4; ++c) {
            const __bf16* kp0 = kb + (size_t)(kbase + lr) * Dc + 32 * c + 8 * half;
            b0[c].u[0] = *(const uint4*)kp0;
            b0[c].u[1] = *(const uint4*)(kp0 + 16);
            const __bf16* kp1 = kb + (size_t)(kbase + 16 + lr) * Dc + 32 * c + 8 * half;
            b1[c].u[0] = *(const uint4*)kp1;
            b1[c].u[1] = *(const uint4*)(kp1 + 16);
        }
        v8f sc0 = {}, sc1 = {};
#pragma unroll
        for (int c = 0; c < 4; ++c) {
            sc0 = wmma_bf16(qf[c], b0[c], sc0);
            sc1 = wmma_bf16(qf[c], b1[c], sc1);
        }

        // ---- online softmax (row stats live per half-wave) ----
        float p0[8], p1[8], alpha[8];
#pragma unroll
        for (int r = 0; r < 8; ++r) {
            int rowg = qt * 16 + r + 8 * half;
            float s0 = sc0[r] * SCALEc + ((kbase + lr) <= rowg ? 0.f : -1e9f);
            float s1 = sc1[r] * SCALEc + ((kbase + 16 + lr) <= rowg ? 0.f : -1e9f);
            float tmax = fmaxf(s0, s1);
#pragma unroll
            for (int off = 8; off > 0; off >>= 1)
                tmax = fmaxf(tmax, __shfl_xor(tmax, off, 32));
            float mnew = fmaxf(m8[r], tmax);
            alpha[r] = __expf(m8[r] - mnew);
            m8[r] = mnew;
            p0[r] = __expf(s0 - mnew);
            p1[r] = __expf(s1 - mnew);
            float rs = p0[r] + p1[r];
#pragma unroll
            for (int off = 8; off > 0; off >>= 1)
                rs += __shfl_xor(rs, off, 32);
            l8[r] = l8[r] * alpha[r] + rs;
        }
#pragma unroll
        for (int n = 0; n < 8; ++n)
#pragma unroll
            for (int r = 0; r < 8; ++r) o[n][r] = o[n][r] * alpha[r];

        // ---- relayout P (16x32) C-layout -> A-layout through LDS ----
        __syncthreads();
#pragma unroll
        for (int r = 0; r < 8; ++r) {
            pl[(r + 8 * half) * 32 + lr]      = (__bf16)p0[r];
            pl[(r + 8 * half) * 32 + 16 + lr] = (__bf16)p1[r];
        }
        __syncthreads();
        Frag pf;
        pf.u[0] = *(const uint4*)(&pl[lr * 32 + 8 * half]);
        pf.u[1] = *(const uint4*)(&pl[lr * 32 + 16 + 8 * half]);

        // ---- PV: two groups of (4 V-frag loads + 4 WMMAs) ----
#pragma unroll
        for (int g = 0; g < 2; ++g) {
            Frag vf[4];
#pragma unroll
            for (int n = 0; n < 4; ++n) {
                const __bf16* vp =
                    vt + (size_t)(16 * (4 * g + n) + lr) * Sc + kbase + 8 * half;
                vf[n].u[0] = *(const uint4*)vp;
                vf[n].u[1] = *(const uint4*)(vp + 16);
            }
#pragma unroll
            for (int n = 0; n < 4; ++n)
                o[4 * g + n] = wmma_bf16(pf, vf[n], o[4 * g + n]);
        }
    }

#pragma unroll
    for (int n = 0; n < 8; ++n)
#pragma unroll
        for (int r = 0; r < 8; ++r) {
            int s = qt * 16 + r + 8 * half;
            float val = o[n][r] / l8[r];
            ctx[((size_t)(b * Sc + s)) * (Hc * Dc) + h * Dc + 16 * n + lr] =
                (__bf16)val;
        }
}

// ---------------------------------------------------------------------------
// Host-side orchestration
// ---------------------------------------------------------------------------
extern "C" void kernel_launch(void* const* d_in, const int* in_sizes, int n_in,
                              void* d_out, int out_size, void* d_ws, size_t ws_size,
                              hipStream_t stream) {
    (void)in_sizes; (void)n_in; (void)out_size; (void)ws_size;

    const float* x    = (const float*)d_in[0];
    const int*   pos  = (const int*)d_in[1];
    // d_in[2] = attn_mask (causal; computed analytically)
    const float* Wq   = (const float*)d_in[3];
    const float* Wk   = (const float*)d_in[4];
    const float* Wv   = (const float*)d_in[5];
    const float* Wo   = (const float*)d_in[6];
    const float* qnw  = (const float*)d_in[7];
    const float* knw  = (const float*)d_in[8];

    constexpr size_t BS = (size_t)Bc * Sc;  // 4096 tokens

    float* out_o = (float*)d_out;                         // [B,S,HID]
    float* out_k = out_o + BS * HIDc;                     // [B,KV,S,D]
    float* out_v = out_k + (size_t)Bc * KVc * Sc * Dc;    // [B,KV,S,D]

    // workspace layout (bytes), all 256-aligned sizes
    char* ws = (char*)d_ws;
    size_t off = 0;
    auto carve = [&](size_t bytes) { char* p = ws + off; off += (bytes + 255) & ~(size_t)255; return p; };

    __bf16* x_bf  = (__bf16*)carve(BS * HIDc * 2);
    __bf16* wq_bf = (__bf16*)carve((size_t)Hc * Dc * HIDc * 2);
    __bf16* wk_bf = (__bf16*)carve((size_t)KVc * Dc * HIDc * 2);
    __bf16* wv_bf = (__bf16*)carve((size_t)KVc * Dc * HIDc * 2);
    __bf16* wo_bf = (__bf16*)carve((size_t)HIDc * Hc * Dc * 2);
    float*  qf32  = (float*)carve(BS * Hc * Dc * 4);
    float*  kf32  = (float*)carve(BS * KVc * Dc * 4);
    float*  vf32  = (float*)carve(BS * KVc * Dc * 4);
    __bf16* q_bf  = (__bf16*)carve((size_t)Bc * Hc * Sc * Dc * 2);
    __bf16* k_bf  = (__bf16*)carve((size_t)Bc * KVc * Sc * Dc * 2);
    __bf16* vT_bf = (__bf16*)carve((size_t)Bc * KVc * Dc * Sc * 2);
    __bf16* ctx_bf = (__bf16*)carve(BS * Hc * Dc * 2);

    // 1) fp32 -> bf16 conversions
    auto cvt = [&](const float* s, __bf16* d, size_t n) {
        int n4 = (int)(n / 4);
        int blocks = (n4 + 255) / 256;
        cvt_f32_bf16<<<blocks, 256, 0, stream>>>((const float4*)s, (ushort4*)d, n4);
    };
    cvt(x,  x_bf,  BS * HIDc);
    cvt(Wq, wq_bf, (size_t)Hc * Dc * HIDc);
    cvt(Wk, wk_bf, (size_t)KVc * Dc * HIDc);
    cvt(Wv, wv_bf, (size_t)KVc * Dc * HIDc);
    cvt(Wo, wo_bf, (size_t)HIDc * Hc * Dc);

    // 2) QKV projections (WMMA GEMM)
    gemm_bf16_nt<<<dim3(BS / GBM, (Hc * Dc) / GBN), 256, 0, stream>>>(
        x_bf, wq_bf, qf32, (int)BS, Hc * Dc, HIDc);
    gemm_bf16_nt<<<dim3(BS / GBM, (KVc * Dc) / GBN), 256, 0, stream>>>(
        x_bf, wk_bf, kf32, (int)BS, KVc * Dc, HIDc);
    gemm_bf16_nt<<<dim3(BS / GBM, (KVc * Dc) / GBN), 256, 0, stream>>>(
        x_bf, wv_bf, vf32, (int)BS, KVc * Dc, HIDc);

    // 3) RMSNorm + RoPE + layout changes
    {
        int waves_q = (int)(BS * Hc);
        normrope_kernel<<<waves_q / 8, 256, 0, stream>>>(
            qf32, pos, qnw, q_bf, nullptr, nullptr, Hc, 1);
        int waves_kv = (int)(BS * KVc);
        normrope_kernel<<<waves_kv / 8, 256, 0, stream>>>(
            kf32, pos, knw, k_bf, out_k, nullptr, KVc, 1);
        normrope_kernel<<<waves_kv / 8, 256, 0, stream>>>(
            vf32, pos, nullptr, nullptr, out_v, vT_bf, KVc, 0);
    }

    // 4) flash attention (WMMA)
    attn_kernel<<<dim3(Sc / 16, Hc, Bc), 32, 0, stream>>>(q_bf, k_bf, vT_bf, ctx_bf);

    // 5) output projection (WMMA GEMM) -> fp32 out
    gemm_bf16_nt<<<dim3(BS / GBM, HIDc / GBN), 256, 0, stream>>>(
        ctx_bf, wo_bf, out_o, (int)BS, HIDc, Hc * Dc);
}
